// GraphAttentionManifold_83537113907342
// MI455X (gfx1250) — matrix operand order repaired and checked by reading, not compiled, not run on hardware
//
#include <hip/hip_runtime.h>

typedef __bf16 bf16_t;
typedef __attribute__((ext_vector_type(16))) __bf16 v16bf;
typedef __attribute__((ext_vector_type(8)))  float  v8f;
typedef __attribute__((ext_vector_type(4)))  unsigned int u32x4;
typedef __attribute__((ext_vector_type(8)))  int i32x8;
typedef __attribute__((ext_vector_type(4)))  int i32x4;

#define B_   2
#define N_   2048
#define D_   512
#define H_   8
#define HD_  64
#define BN_  (B_ * N_)
#define EPS_ 1e-9f

union frag_u { uint4 q[2]; v16bf v; };
union pk8_u  { uint4 u; bf16_t h[8]; };

// =====================================================================
// TDM: DMA a 64x64 bf16 2D tile (row stride = stride_elems) into LDS,
// padding each 128B row by 16B -> 144B LDS stride (72 bf16).
// D# fields per CDNA5 ISA.
// =====================================================================
__device__ static inline void tdm_load_tile64(unsigned lds_addr, const void* gptr,
                                              unsigned stride_elems)
{
    unsigned long long ga = (unsigned long long)(size_t)gptr;
    u32x4 g0;
    g0[0] = 1u;                                               // count=1 (valid), user mode
    g0[1] = lds_addr;                                         // lds_addr
    g0[2] = (unsigned)(ga & 0xFFFFFFFFu);                     // global_addr[31:0]
    g0[3] = (unsigned)((ga >> 32) & 0x1FFFFFFu) | (2u << 30); // global_addr[56:32], type=2
    i32x8 g1;
    g1[0] = (int)((1u << 16)      // data_size = 2 bytes
                | (1u << 20)      // pad_enable
                | (4u << 22)      // pad_interval: 32 dwords (128B)
                | (3u << 25));    // pad_amount: 4 dwords (16B)
    g1[1] = (int)(64u << 16);     // tensor_dim0 = 64
    g1[2] = (int)(64u << 16);     // tensor_dim1 = 64
    g1[3] = (int)(64u << 16);     // tile_dim0   = 64
    g1[4] = (int)64;              // tile_dim1   = 64
    g1[5] = (int)stride_elems;    // tensor_dim0_stride[31:0]
    g1[6] = 0;
    g1[7] = 0;
    i32x4 gz = {0, 0, 0, 0};
#if __clang_major__ >= 23
    i32x8 gz8 = {0, 0, 0, 0, 0, 0, 0, 0};
    __builtin_amdgcn_tensor_load_to_lds(g0, g1, gz, gz, gz8, 0);
#else
    __builtin_amdgcn_tensor_load_to_lds(g0, g1, gz, gz, 0);
#endif
}

// =====================================================================
// QKV projection: out = bf16(X @ W + bias); TR=true stores out transposed
// ([n][m], for V so attention V-tiles are row-contiguous).
// block tile 64x64, 8 waves = 4(M) x 2(N) sub-tiles of 16x32.
// =====================================================================
template <bool TR>
__global__ __launch_bounds__(256) void qkv_gemm_kernel(
        const float* __restrict__ X, const float* __restrict__ W,
        const float* __restrict__ bias, bf16_t* __restrict__ out)
{
    __shared__ bf16_t sX [64][40];  // [m][k]
    __shared__ bf16_t sWt[64][40];  // [n][k] transposed

    const int tid = threadIdx.x;
    const int wv  = tid >> 5;
    const int ln  = tid & 31;
    const int l16 = ln & 15;
    const bool lo = (ln < 16);

    const int m0 = blockIdx.y * 64;
    const int n0 = blockIdx.x * 64;
    const int mi = (wv & 3) * 16;
    const int ni = (wv >> 2) * 32;

    v8f acc[2] = {};

    for (int k0 = 0; k0 < D_; k0 += 32) {
        __syncthreads();
        {   // stage X tile (64x32) as bf16, vectorized b128 store
            const int row = tid >> 2;
            const int ch  = (tid & 3) * 8;
            const float* gp = X + (size_t)(m0 + row) * D_ + k0 + ch;
            union { float4 v; float f[4]; } u0, u1;
            u0.v = *(const float4*)gp;
            u1.v = *(const float4*)(gp + 4);
            pk8_u pk;
            #pragma unroll
            for (int e = 0; e < 4; ++e) { pk.h[e] = (bf16_t)u0.f[e]; pk.h[4 + e] = (bf16_t)u1.f[e]; }
            *(uint4*)&sX[row][ch] = pk.u;
        }
        {   // stage W tile (32x64) transposed -> sWt[n][k]
            const int r  = tid >> 3;
            const int ch = (tid & 7) * 8;
            const float* gp = W + (size_t)(k0 + r) * D_ + n0 + ch;
            union { float4 v; float f[4]; } u0, u1;
            u0.v = *(const float4*)gp;
            u1.v = *(const float4*)(gp + 4);
            #pragma unroll
            for (int e = 0; e < 4; ++e) {
                sWt[ch + e    ][r] = (bf16_t)u0.f[e];
                sWt[ch + e + 4][r] = (bf16_t)u1.f[e];
            }
        }
        __syncthreads();

        frag_u afr;  // A 16x32: lane<16 -> K {0..7,16..23}; lane>=16 -> {8..15,24..31}
        {
            const int row  = mi + l16;
            const int base = lo ? 0 : 8;
            afr.q[0] = *(const uint4*)&sX[row][base];
            afr.q[1] = *(const uint4*)&sX[row][base + 16];
        }
        #pragma unroll
        for (int ns = 0; ns < 2; ++ns) {
            frag_u bfr;  // B 32x16: lane<16 -> N=l16, K {0..7,16..23}; lane>=16 -> K {8..15,24..31}
            const int n    = ni + ns * 16 + l16;
            const int koff = lo ? 0 : 8;
            bfr.q[0] = *(const uint4*)&sWt[n][koff];
            bfr.q[1] = *(const uint4*)&sWt[n][koff + 16];
            acc[ns] = __builtin_amdgcn_wmma_f32_16x16x32_bf16(
                    false, afr.v, false, bfr.v, (short)0, acc[ns], false, false);
        }
    }

    // epilogue (D-layout: VGPR r -> rows {r, 8+r} in lane halves, N = l16)
    #pragma unroll
    for (int ns = 0; ns < 2; ++ns) {
        const int n = n0 + ni + ns * 16 + l16;
        const float bv = bias[n];
        if (TR) {
            pk8_u pk;
            #pragma unroll
            for (int r = 0; r < 8; ++r) pk.h[r] = (bf16_t)(acc[ns][r] + bv);
            const int row0 = m0 + mi + (lo ? 0 : 8);
            *(uint4*)&out[(size_t)n * BN_ + row0] = pk.u;   // 8 contiguous rows
        } else {
            #pragma unroll
            for (int r = 0; r < 8; ++r) {
                const int row = m0 + mi + (lo ? r : 8 + r);
                out[(size_t)row * D_ + n] = (bf16_t)(acc[ns][r] + bv);
            }
        }
    }
}

// =====================================================================
// gate: a[row] = x[row].w_g[:D], c[row] = x[row].w_g[D:]
// =====================================================================
__global__ __launch_bounds__(256) void gate_kernel(
        const float* __restrict__ X, const float* __restrict__ wg,
        float* __restrict__ ga, float* __restrict__ gc)
{
    const int row = blockIdx.x * 8 + (threadIdx.x >> 5);
    const int ln  = threadIdx.x & 31;
    const float* xp = X + (size_t)row * D_;
    float sa = 0.0f, sc = 0.0f;
    for (int k = ln; k < D_; k += 32) {
        const float xv = xp[k];
        sa += xv * wg[k];
        sc += xv * wg[D_ + k];
    }
    #pragma unroll
    for (int msk = 16; msk >= 1; msk >>= 1) {
        sa += __shfl_xor(sa, msk);
        sc += __shfl_xor(sc, msk);
    }
    if (ln == 0) { ga[row] = sa; gc[row] = sc; }
}

// =====================================================================
// bias plane: bias[b,i,j] = a_i + c_j + b_g + log(adj+eps)  (head-independent)
// =====================================================================
__global__ __launch_bounds__(256) void bias_kernel(
        const float* __restrict__ adj, const float* __restrict__ ga,
        const float* __restrict__ gc, const float* __restrict__ bg,
        float* __restrict__ biasws)
{
    const int bi = blockIdx.x;          // b*N + i
    const int b  = bi / N_;
    const float av = ga[bi] + bg[0];
    const float* arow = adj + (size_t)bi * N_;
    const float* crow = gc + (size_t)b * N_;
    float* orow = biasws + (size_t)bi * N_;
    for (int j = threadIdx.x; j < N_; j += 256)
        orow[j] = av + crow[j] + __logf(arow[j] + EPS_);
}

// =====================================================================
// flash attention: block = (128 i-rows, head, batch); wave w owns 16 rows.
// K/V tiles double-buffered in LDS via TDM (DMA of tile t+1 overlaps
// compute of tile t); 8+8 wmma per 64-wide j tile.
// =====================================================================
__global__ __launch_bounds__(256) void attn_kernel(
        const bf16_t* __restrict__ Qb, const bf16_t* __restrict__ Kb,
        const bf16_t* __restrict__ Vt, const float* __restrict__ biasws,
        const float* __restrict__ X, float* __restrict__ out)
{
    __shared__ bf16_t sK [2][64][72];   // [buf][j][hd]  (TDM, 144B padded rows)
    __shared__ bf16_t sVt[2][64][72];   // [buf][hd][j]  (TDM, V pre-transposed in ws)
    __shared__ bf16_t sP [8][16][72];   // wave-private D->A relayout scratch

    const int tid = threadIdx.x;
    const int wv  = tid >> 5;
    const int ln  = tid & 31;
    const int l16 = ln & 15;
    const bool lo = (ln < 16);

    const int h  = blockIdx.y;
    const int b  = blockIdx.z;
    const int i0 = blockIdx.x * 128 + wv * 16;
    const size_t rowbase = (size_t)b * N_;

    // Q fragments resident in VGPRs (two 16x32 A-frags over hd)
    frag_u qa[2];
    {
        const bf16_t* qp = Qb + (rowbase + i0 + l16) * D_ + h * HD_;
        const int base = lo ? 0 : 8;
        #pragma unroll
        for (int kc = 0; kc < 2; ++kc) {
            qa[kc].q[0] = *(const uint4*)(qp + kc * 32 + base);
            qa[kc].q[1] = *(const uint4*)(qp + kc * 32 + base + 16);
        }
    }

    float rm[8], rl[8];
    #pragma unroll
    for (int r = 0; r < 8; ++r) { rm[r] = -3.0e38f; rl[r] = 0.0f; }
    v8f acc[4] = {};
    const float scale = 0.125f;  // 1/sqrt(64)

    const unsigned ldsKa[2] = { (unsigned)(size_t)&sK [0][0][0], (unsigned)(size_t)&sK [1][0][0] };
    const unsigned ldsVa[2] = { (unsigned)(size_t)&sVt[0][0][0], (unsigned)(size_t)&sVt[1][0][0] };
    const bf16_t* Kbase = Kb + rowbase * D_ + h * HD_;
    const bf16_t* Vbase = Vt + (size_t)(h * HD_) * BN_ + rowbase;

    // prologue: kick off DMA of tile 0 into buffer 0
    if (wv == 0) {
        tdm_load_tile64(ldsKa[0], Kbase, D_);
        tdm_load_tile64(ldsVa[0], Vbase, BN_);
    }

    for (int j0 = 0; j0 < N_; j0 += 64) {
        const int buf = (j0 >> 6) & 1;
        if (wv == 0) __builtin_amdgcn_s_wait_tensorcnt(0);
        __syncthreads();   // tile[buf] published; tile[buf^1] fully consumed

        // DMA next tile while this one is being computed on
        if (wv == 0 && j0 + 64 < N_) {
            tdm_load_tile64(ldsKa[buf ^ 1], Kbase + (size_t)(j0 + 64) * D_, D_);
            tdm_load_tile64(ldsVa[buf ^ 1], Vbase + j0 + 64, BN_);
        }
        if (j0 + 64 < N_)  // global_prefetch_b8 next bias tile rows
            __builtin_prefetch(biasws + (rowbase + i0 + l16) * (size_t)N_ + j0 + 64, 0, 1);

        // hoist bias tile loads so they overlap the S wmmas
        float bs[4][8];
        #pragma unroll
        for (int js = 0; js < 4; ++js)
            #pragma unroll
            for (int r = 0; r < 8; ++r) {
                const int row = i0 + (lo ? r : 8 + r);
                bs[js][r] = biasws[(rowbase + row) * (size_t)N_ + j0 + js * 16 + l16];
            }

        // ---- S = Q K^T, 4 sub-tiles of 16x16 over j ----
        v8f sacc[4] = {};
        #pragma unroll
        for (int kc = 0; kc < 2; ++kc) {     // A held constant across inner wmma
            #pragma unroll
            for (int js = 0; js < 4; ++js) {
                frag_u bfr;  // B = K^T: (kk, j) = K[j][kk]
                const int j    = js * 16 + l16;
                const int koff = kc * 32 + (lo ? 0 : 8);
                bfr.q[0] = *(const uint4*)&sK[buf][j][koff];
                bfr.q[1] = *(const uint4*)&sK[buf][j][koff + 16];
                sacc[js] = __builtin_amdgcn_wmma_f32_16x16x32_bf16(
                        false, qa[kc].v, false, bfr.v, (short)0, sacc[js], false, false);
            }
        }
        float s[4][8];
        #pragma unroll
        for (int js = 0; js < 4; ++js)
            #pragma unroll
            for (int r = 0; r < 8; ++r)
                s[js][r] = sacc[js][r] * scale + bs[js][r];

        // ---- online softmax (rows in 16-lane halves; xor<16 stays in half) ----
        #pragma unroll
        for (int r = 0; r < 8; ++r) {
            float mx = fmaxf(fmaxf(s[0][r], s[1][r]), fmaxf(s[2][r], s[3][r]));
            #pragma unroll
            for (int msk = 8; msk >= 1; msk >>= 1) mx = fmaxf(mx, __shfl_xor(mx, msk));
            const float mn   = fmaxf(rm[r], mx);
            const float corr = __expf(rm[r] - mn);
            rm[r] = mn;
            float rs = 0.0f;
            #pragma unroll
            for (int js = 0; js < 4; ++js) {
                const float e = __expf(s[js][r] - mn);
                s[js][r] = e;
                rs += e;
            }
            #pragma unroll
            for (int msk = 8; msk >= 1; msk >>= 1) rs += __shfl_xor(rs, msk);
            rl[r] = rl[r] * corr + rs;
            #pragma unroll
            for (int cn = 0; cn < 4; ++cn) acc[cn][r] *= corr;
        }

        // ---- P: D-layout -> A-layout via wave-private LDS (same-wave DS in order) ----
        #pragma unroll
        for (int js = 0; js < 4; ++js)
            #pragma unroll
            for (int r = 0; r < 8; ++r)
                sP[wv][lo ? r : 8 + r][js * 16 + l16] = (bf16_t)s[js][r];

        frag_u pa[2];
        #pragma unroll
        for (int kc = 0; kc < 2; ++kc) {
            const int base = kc * 32 + (lo ? 0 : 8);
            pa[kc].q[0] = *(const uint4*)&sP[wv][l16][base];
            pa[kc].q[1] = *(const uint4*)&sP[wv][l16][base + 16];
        }

        // ---- acc += P V ----
        #pragma unroll
        for (int kc = 0; kc < 2; ++kc) {     // A held constant across inner wmma
            #pragma unroll
            for (int cn = 0; cn < 4; ++cn) {
                frag_u bfr;  // B = V: (j, c) = sVt[c][j]
                const int c    = cn * 16 + l16;
                const int joff = kc * 32 + (lo ? 0 : 8);
                bfr.q[0] = *(const uint4*)&sVt[buf][c][joff];
                bfr.q[1] = *(const uint4*)&sVt[buf][c][joff + 16];
                acc[cn] = __builtin_amdgcn_wmma_f32_16x16x32_bf16(
                        false, pa[kc].v, false, bfr.v, (short)0, acc[cn], false, false);
            }
        }
    }

    // ---- epilogue: out = acc / l + x (residual) ----
    #pragma unroll
    for (int cn = 0; cn < 4; ++cn) {
        #pragma unroll
        for (int r = 0; r < 8; ++r) {
            const int row = i0 + (lo ? r : 8 + r);
            const int col = h * HD_ + cn * 16 + l16;
            const size_t idx = (rowbase + row) * (size_t)D_ + col;
            out[idx] = acc[cn][r] / rl[r] + X[idx];
        }
    }
}

// =====================================================================
extern "C" void kernel_launch(void* const* d_in, const int* in_sizes, int n_in,
                              void* d_out, int out_size, void* d_ws, size_t ws_size,
                              hipStream_t stream)
{
    (void)in_sizes; (void)n_in; (void)out_size; (void)ws_size;
    const float* x   = (const float*)d_in[0];
    const float* adj = (const float*)d_in[1];
    const float* Wq  = (const float*)d_in[2];
    const float* bq  = (const float*)d_in[3];
    const float* Wk  = (const float*)d_in[4];
    const float* bk  = (const float*)d_in[5];
    const float* Wv  = (const float*)d_in[6];
    const float* bv  = (const float*)d_in[7];
    const float* wg  = (const float*)d_in[8];
    const float* bg  = (const float*)d_in[9];
    float* out = (float*)d_out;

    char* ws = (char*)d_ws;
    bf16_t* Qb = (bf16_t*)(ws);
    bf16_t* Kb = (bf16_t*)(ws + (size_t)4 * 1024 * 1024);
    bf16_t* Vt = (bf16_t*)(ws + (size_t)8 * 1024 * 1024);  // transposed: [col][b*N+row]
    float*  ga = (float*)(ws + (size_t)12 * 1024 * 1024);
    float*  gc = ga + BN_;
    float*  biasws = (float*)(ws + (size_t)12 * 1024 * 1024 + 65536);

    dim3 gq(8, 64);
    qkv_gemm_kernel<false><<<gq, 256, 0, stream>>>(x, Wq, bq, Qb);
    qkv_gemm_kernel<false><<<gq, 256, 0, stream>>>(x, Wk, bk, Kb);
    qkv_gemm_kernel<true ><<<gq, 256, 0, stream>>>(x, Wv, bv, Vt);
    gate_kernel<<<BN_ / 8, 256, 0, stream>>>(x, wg, ga, gc);
    bias_kernel<<<B_ * N_, 256, 0, stream>>>(adj, ga, gc, bg, biasws);
    dim3 gat(N_ / 128, H_, B_);
    attn_kernel<<<gat, 256, 0, stream>>>(Qb, Kb, Vt, biasws, x, out);
}
